// ConvBlock_38517266711178
// MI455X (gfx1250) — matrix operand order repaired and checked by reading, not compiled, not run on hardware
//
#include <hip/hip_runtime.h>

typedef _Float16 v8h  __attribute__((ext_vector_type(8)));
typedef _Float16 v16h __attribute__((ext_vector_type(16)));
typedef float    v8f  __attribute__((ext_vector_type(8)));

// Problem geometry (fixed by the reference)
constexpr int Bn   = 64;      // batch
constexpr int Cin  = 64;      // input channels
constexpr int Himg = 32, Wimg = 32;
constexpr int Lpix = Himg * Wimg;        // 1024 pixels / image
constexpr int OUTC = 64;                 // p*mb
constexpr int Kdim = Cin * 9;            // 576 = q*mb
constexpr int Mtot = Bn * Lpix;          // 65536 GEMM rows
constexpr int MTILE = 128;               // pixels per workgroup (4 image rows)
constexpr int NWG   = Mtot / MTILE;      // 512 workgroups

// MORR constants
constexpr float A_MORR = 0.8578f;
constexpr float R_MORR = 0.8578f;
constexpr float MC1 = A_MORR * A_MORR + R_MORR * R_MORR;     // num: MC1 - MC2*cos
constexpr float MC2 = 2.0f * A_MORR * R_MORR;                // shared cos coefficient
constexpr float MC3 = 1.0f + (A_MORR * R_MORR) * (A_MORR * R_MORR); // den: MC3 - MC2*cos

// ---------------------------------------------------------------------------
// Kernel 0: expand circulant weight (p,q,mb) -> Wfull^T [n=outC][k=576] in f16
//   Wfull[k=q*8+j][n=p*8+i] = weight[p, q, (i-j) mod 8]
// ---------------------------------------------------------------------------
__global__ void expand_weights(const float* __restrict__ w, _Float16* __restrict__ wexp) {
    int idx = blockIdx.x * blockDim.x + threadIdx.x;   // 64*576 = 36864 threads
    if (idx >= OUTC * Kdim) return;
    int n = idx / Kdim, k = idx - n * Kdim;
    int p = n >> 3, i = n & 7;
    int q = k >> 3, j = k & 7;
    float v = w[(p * (Kdim / 8) + q) * 8 + ((i - j) & 7)];
    wexp[idx] = (_Float16)v;
}

// Kernel 0b: zero the 128 channel accumulators (sum, sumsq)
__global__ void zero_sums(float* __restrict__ g) { g[threadIdx.x] = 0.0f; }

// ---------------------------------------------------------------------------
// Kernel 1: fused im2col^2 + WMMA GEMM + MORR transfer + channel-sum reduce.
// 256 threads (8 wave32). LDS: S-tile 128x576 f16 + W^T 64x576 f16 (dynamic).
// ---------------------------------------------------------------------------
__global__ void morr_gemm(const float* __restrict__ x,
                          const _Float16* __restrict__ wexp,
                          float* __restrict__ out,
                          float* __restrict__ gsum,
                          float* __restrict__ gsq) {
    extern __shared__ char smem[];
    _Float16* Slds = (_Float16*)smem;                         // 128*576 f16
    _Float16* Wlds = (_Float16*)(smem + MTILE * Kdim * 2);    //  64*576 f16
    __shared__ float csum[OUTC];
    __shared__ float cssq[OUTC];

    const int t = threadIdx.x;
    const int g = blockIdx.x;
    if (t < OUTC) { csum[t] = 0.0f; cssq[t] = 0.0f; }

    // ---- stage W^T into LDS (16B vector copies) ----
    {
        const uint4* src = (const uint4*)wexp;
        uint4*       dst = (uint4*)Wlds;
        const int nvec = (OUTC * Kdim * 2) / 16;              // 4608
        for (int i = t; i < nvec; i += 256) dst[i] = src[i];
    }

    // ---- stage squared im2col tile: 128 pixels x 576 ----
    const int P0 = g * MTILE;                 // first global pixel row
    const int b  = P0 >> 10;                  // whole tile lives in one image
    const float* xb = x + (size_t)b * (Cin * Lpix);
    {
        const int m  = t >> 1;                // pixel within tile (2 threads/pixel)
        const int c0 = (t & 1) * (Cin / 2);   // each thread: 32 channels
        const int l  = (P0 & (Lpix - 1)) + m;
        const int h  = l >> 5, w0 = l & 31;
        _Float16* srow = Slds + m * Kdim;
        for (int c = c0; c < c0 + Cin / 2; ++c) {
            const float* xc = xb + c * Lpix;
            int k = c * 9;
#pragma unroll
            for (int di = 0; di < 3; ++di) {
                const int  hh  = h + di - 1;
                const bool okh = (hh >= 0) & (hh < Himg);
#pragma unroll
                for (int dj = 0; dj < 3; ++dj, ++k) {
                    const int ww = w0 + dj - 1;
                    float v = (okh & (ww >= 0) & (ww < Wimg)) ? xc[hh * Wimg + ww] : 0.0f;
                    srow[k] = (_Float16)(v * v);
                }
            }
        }
    }
    __syncthreads();

    // ---- WMMA: each wave -> 16 rows x 64 cols, K = 576 in 18 steps of 32 ----
    const int wave = t >> 5;
    const int lane = t & 31;
    const int row  = lane & 15;
    const int half = lane >> 4;
    const int m0   = wave * 16;

    v8f acc[4] = {v8f{}, v8f{}, v8f{}, v8f{}};
    const char* abase = (const char*)(Slds + (m0 + row) * Kdim);

    for (int kk = 0; kk < Kdim / 32; ++kk) {
        // A fragment (16x32): this code's contiguous K half-split order
        //   elems 0-7  : K = kk*32 + half*16 + (0..7)
        //   elems 8-15 : K = kk*32 + half*16 + 8 + (0..7)
        union { v16h v; v8h h[2]; } afrag;
        const char* ap = abase + kk * 64 + half * 32;
        afrag.h[0] = *(const v8h*)(ap);
        afrag.h[1] = *(const v8h*)(ap + 16);
#pragma unroll
        for (int tn = 0; tn < 4; ++tn) {
            // B fragment (32x16): lane = column n, contiguous K half-split
            //   lanes 0-15: K = kk*32 + 0..15 ; lanes 16-31: K = kk*32 + 16..31
            const int n = tn * 16 + row;
            const v16h bfrag =
                *(const v16h*)((const char*)(Wlds + n * Kdim) + kk * 64 + half * 32);
            acc[tn] = __builtin_amdgcn_wmma_f32_16x16x32_f16(
                false, afrag.v, false, bfrag, (short)0, acc[tn], false, false);
        }
    }

    // ---- MORR transfer + store (B,C,H,W layout) + per-channel reduction ----
    const int lbase = (P0 & (Lpix - 1));
#pragma unroll
    for (int tn = 0; tn < 4; ++tn) {
        const int n = tn * 16 + row;
        float s = 0.0f, sq = 0.0f;
#pragma unroll
        for (int r = 0; r < 8; ++r) {
            const float phi = acc[tn][r];
            const float cp  = cosf(phi);
            const float tr  = (MC1 - MC2 * cp) / (MC3 - MC2 * cp);
            const int   m   = m0 + r + 8 * half;           // C-layout: M = r (+8 hi half)
            out[((size_t)b * OUTC + n) * Lpix + lbase + m] = tr;
            s  += tr;
            sq += tr * tr;
        }
        s  += __shfl_xor(s, 16, 32);
        sq += __shfl_xor(sq, 16, 32);
        if (half == 0) {
            atomicAdd(&csum[n], s);       // ds_add_f32
            atomicAdd(&cssq[n], sq);
        }
    }
    __syncthreads();
    if (t < OUTC)            atomicAdd(&gsum[t], csum[t]);
    else if (t < 2 * OUTC)   atomicAdd(&gsq[t - OUTC], cssq[t - OUTC]);
}

// ---------------------------------------------------------------------------
// Kernel 2: BatchNorm apply + clip, in place on d_out (B, C, H, W).
// ---------------------------------------------------------------------------
__global__ void bn_apply(float* __restrict__ out,
                         const float* __restrict__ gsum,
                         const float* __restrict__ gsq,
                         const float* __restrict__ gamma,
                         const float* __restrict__ beta) {
    const int idx = blockIdx.x * blockDim.x + threadIdx.x;   // 4194304 elements
    const int n   = (idx >> 10) & (OUTC - 1);                // channel
    const float invN = 1.0f / (float)(Bn * Lpix);            // 1/65536
    const float mean = gsum[n] * invN;
    const float var  = gsq[n] * invN - mean * mean;
    const float rs   = rsqrtf(var + 1e-5f);
    const float v    = out[idx];
    float y = (v - mean) * rs * gamma[n] + beta[n];
    y = fminf(1.0f, fmaxf(-1.0f, y));
    out[idx] = y;
}

// ---------------------------------------------------------------------------
extern "C" void kernel_launch(void* const* d_in, const int* in_sizes, int n_in,
                              void* d_out, int out_size, void* d_ws, size_t ws_size,
                              hipStream_t stream) {
    const float* x      = (const float*)d_in[0];   // (64,64,32,32)
    const float* weight = (const float*)d_in[1];   // (8,72,8)
    const float* gamma  = (const float*)d_in[2];   // (64,)
    const float* beta   = (const float*)d_in[3];   // (64,)
    float*       out    = (float*)d_out;           // (64,64,32,32)

    // workspace layout: [0,73728) Wfull^T f16 ; then 64+64 floats of channel sums
    char*      ws   = (char*)d_ws;
    _Float16*  wexp = (_Float16*)ws;
    float*     gsum = (float*)(ws + (size_t)OUTC * Kdim * 2);
    float*     gsq  = gsum + OUTC;

    expand_weights<<<(OUTC * Kdim + 255) / 256, 256, 0, stream>>>(weight, wexp);
    zero_sums<<<1, 2 * OUTC, 0, stream>>>(gsum);

    const size_t smem = (size_t)(MTILE + OUTC) * Kdim * 2;   // 221184 B < 320 KB WGP LDS
    morr_gemm<<<NWG, 256, smem, stream>>>(x, wexp, out, gsum, gsq);

    bn_apply<<<(Bn * OUTC * Lpix) / 256, 256, 0, stream>>>(out, gsum, gsq, gamma, beta);
}
